// TriangleMultiplicationOutgoing_2199023255610
// MI455X (gfx1250) — compile-verified, matrix operand order checked by reading, not run on hardware
//
#include <hip/hip_runtime.h>

#define NDIM 384
#define CDIM 128
#define NN (384 * 384)
#define LDA 136   // bf16 pitch: 136*2 = 272 bytes, 16B aligned rows

typedef __attribute__((ext_vector_type(16))) __bf16   v16bf;
typedef __attribute__((ext_vector_type(8)))  float    v8f;
typedef __attribute__((ext_vector_type(4)))  unsigned u32x4;
typedef __attribute__((ext_vector_type(4)))  float    f32x4;

// ---------- small helpers ----------
__device__ inline unsigned short f2bfbits(float f) {
  unsigned u = __builtin_bit_cast(unsigned, f);
  u += 0x7FFFu + ((u >> 16) & 1u);          // round-to-nearest-even
  return (unsigned short)(u >> 16);
}
__device__ inline float bf2f(unsigned short b) {
  return __builtin_bit_cast(float, (unsigned)b << 16);
}
__device__ inline float sigmoidf(float x) { return 1.0f / (1.0f + __expf(-x)); }

__device__ inline v8f zero8() {
  v8f z;
  #pragma unroll
  for (int i = 0; i < 8; i++) z[i] = 0.0f;
  return z;
}

__device__ inline void lds_fence() { asm volatile("s_wait_dscnt 0" ::: "memory"); }

// Load a 16-element bf16 fragment as two contiguous b128 chunks.
// Element e<8  -> k = e  + 8*(lane>>4)   (first  run, base+0)
// Element e>=8 -> k = 8+e+ 8*(lane>>4)   (second run, base+16)
__device__ inline v16bf load_frag(const unsigned short* p) {
  union { u32x4 u[2]; v16bf v; } f;
  f.u[0] = *(const u32x4*)(p);
  f.u[1] = *(const u32x4*)(p + 16);
  return f.v;
}

__device__ inline v8f wmma_bf16(v16bf a, v16bf b, v8f c) {
  return __builtin_amdgcn_wmma_f32_16x16x32_bf16(false, a, false, b, (short)0, c,
                                                 false, false);
}

// 4-lane (adjacent) reduction within wave32
__device__ inline float red4(float v) {
  v += __shfl_xor(v, 1, 32);
  v += __shfl_xor(v, 2, 32);
  return v;
}

// Stage a 128x128 fp32 weight into LDS as bf16 with pitch LDA
__device__ inline void stage_w(const float* __restrict__ w, unsigned short* wl, int tid) {
  __syncthreads();
  for (int idx = tid; idx < CDIM * CDIM; idx += 256)
    wl[(idx >> 7) * LDA + (idx & 127)] = f2bfbits(w[idx]);
  __syncthreads();
}

// 64x128 (A in LDS) x 128x128 (W in LDS, stored [n][k]) -> wave's 4 tiles
__device__ inline void gemm4(const unsigned short* A, const unsigned short* W,
                             int mt, int nh, int lane, v8f acc[4]) {
  const int ml = lane & 15, h = lane >> 4;
  #pragma unroll
  for (int kk = 0; kk < 4; kk++) {
    v16bf a = load_frag(A + (mt * 16 + ml) * LDA + kk * 32 + 8 * h);
    #pragma unroll
    for (int j = 0; j < 4; j++) {
      v16bf b = load_frag(W + ((nh * 4 + j) * 16 + ml) * LDA + kk * 32 + 8 * h);
      acc[j] = wmma_bf16(a, b, acc[j]);
    }
  }
}

// ===================== Kernel 1: LN + 5 projections + gates =====================
__global__ __launch_bounds__(256) void k1_proj(
    const float* __restrict__ x,   const float* __restrict__ n1g, const float* __restrict__ n1b,
    const float* __restrict__ w1i, const float* __restrict__ b1i,
    const float* __restrict__ w1j, const float* __restrict__ b1j,
    const float* __restrict__ w1is,const float* __restrict__ b1is,
    const float* __restrict__ w1js,const float* __restrict__ b1js,
    const float* __restrict__ w3,  const float* __restrict__ b3,
    unsigned short* __restrict__ i_ws, unsigned short* __restrict__ j_ws,
    unsigned short* __restrict__ g3_ws)
{
  __shared__ unsigned short xln[64 * LDA];
  __shared__ unsigned short wlds[128 * LDA];
  __shared__ unsigned short tbuf[8 * 16 * 17];

  const int tid = threadIdx.x;
  const int R0  = blockIdx.x * 64;

  // ---- LayerNorm 64 rows: 4 threads/row, 32 ch each ----
  {
    const int r = tid >> 2, q = tid & 3;
    const float* xr = x + (size_t)(R0 + r) * CDIM + q * 32;
    float v[32];
    float s = 0.f, s2 = 0.f;
    #pragma unroll
    for (int u = 0; u < 8; u++) {
      f32x4 t = ((const f32x4*)xr)[u];
      v[u*4+0] = t.x; v[u*4+1] = t.y; v[u*4+2] = t.z; v[u*4+3] = t.w;
      s  += t.x + t.y + t.z + t.w;
      s2 += t.x*t.x + t.y*t.y + t.z*t.z + t.w*t.w;
    }
    s = red4(s); s2 = red4(s2);
    const float mean = s * (1.0f / 128.0f);
    const float var  = s2 * (1.0f / 128.0f) - mean * mean;
    const float rstd = rsqrtf(var + 1e-5f);
    #pragma unroll
    for (int u = 0; u < 32; u++) {
      int c = q * 32 + u;
      float y = (v[u] - mean) * rstd * n1g[c] + n1b[c];
      xln[r * LDA + c] = f2bfbits(y);
    }
  }
  __syncthreads();

  const int w = tid >> 5, lane = tid & 31;
  const int mt = w & 3, nh = w >> 2;
  const int ml = lane & 15, h = lane >> 4;
  unsigned short* tb = tbuf + w * 272;

  v8f pa[4], ps[4];

  // ---- helper lambda: gate, transpose through LDS, store channel-major ----
  auto gate_store = [&](const float* __restrict__ bp, const float* __restrict__ bs,
                        unsigned short* __restrict__ dst) {
    #pragma unroll
    for (int j = 0; j < 4; j++) {
      const int ng = (nh * 4 + j) * 16 + ml;
      const float bi = bp[ng], bis = bs[ng];
      #pragma unroll
      for (int g = 0; g < 8; g++) {
        float val = (pa[j][g] + bi) * sigmoidf(ps[j][g] + bis);
        tb[(g + 8 * h) * 17 + ml] = f2bfbits(val);   // tbuf[m][n]
      }
      lds_fence();
      #pragma unroll
      for (int p = 0; p < 8; p++) {
        int nl = 2 * p + h;
        int c  = (nh * 4 + j) * 16 + nl;
        dst[(size_t)c * NN + R0 + mt * 16 + ml] = tb[ml * 17 + nl];
      }
      lds_fence();
    }
  };

  // ---- i = (x@w1i^T + b1i) * sigmoid(x@w1i_s^T + b1i_s) ----
  stage_w(w1i, wlds, tid);
  #pragma unroll
  for (int j = 0; j < 4; j++) pa[j] = zero8();
  gemm4(xln, wlds, mt, nh, lane, pa);
  stage_w(w1is, wlds, tid);
  #pragma unroll
  for (int j = 0; j < 4; j++) ps[j] = zero8();
  gemm4(xln, wlds, mt, nh, lane, ps);
  gate_store(b1i, b1is, i_ws);

  // ---- j path ----
  stage_w(w1j, wlds, tid);
  #pragma unroll
  for (int j = 0; j < 4; j++) pa[j] = zero8();
  gemm4(xln, wlds, mt, nh, lane, pa);
  stage_w(w1js, wlds, tid);
  #pragma unroll
  for (int j = 0; j < 4; j++) ps[j] = zero8();
  gemm4(xln, wlds, mt, nh, lane, ps);
  gate_store(b1j, b1js, j_ws);

  // ---- g3 = sigmoid(x@w3^T + b3), stored row-major [R][c] ----
  stage_w(w3, wlds, tid);
  #pragma unroll
  for (int j = 0; j < 4; j++) pa[j] = zero8();
  gemm4(xln, wlds, mt, nh, lane, pa);
  #pragma unroll
  for (int j = 0; j < 4; j++) {
    const int ng = (nh * 4 + j) * 16 + ml;
    const float bb = b3[ng];
    #pragma unroll
    for (int g = 0; g < 8; g++) {
      int R = R0 + mt * 16 + g + 8 * h;
      g3_ws[(size_t)R * CDIM + ng] = f2bfbits(sigmoidf(pa[j][g] + bb));
    }
  }
}

// ===================== Kernel 2: per-channel triangle einsum =====================
// out[c] = I_c (MxK row-major) @ J_c^T ; j_ws[c] is [n][k] == B column-major,
// so both fragments are straight contiguous b128 loads from global.
__global__ __launch_bounds__(256) void k2_tri(const unsigned short* __restrict__ iw,
                                              const unsigned short* __restrict__ jw,
                                              unsigned short* __restrict__ op)
{
  const int bx = blockIdx.x;
  const int c  = bx / 9, t = bx % 9;
  const int tm = (t / 3) * 128, tn = (t % 3) * 128;
  const int w = threadIdx.x >> 5, lane = threadIdx.x & 31;
  const int m0 = tm + (w >> 1) * 32, n0 = tn + (w & 1) * 64;
  const int ml = lane & 15, h = lane >> 4;

  const unsigned short* A = iw + (size_t)c * NN;
  const unsigned short* B = jw + (size_t)c * NN;

  v8f acc[2][4];
  #pragma unroll
  for (int mi = 0; mi < 2; mi++)
    #pragma unroll
    for (int j = 0; j < 4; j++) acc[mi][j] = zero8();

  for (int kk = 0; kk < NDIM; kk += 32) {
    __builtin_prefetch(A + (size_t)(m0 + ml) * NDIM + kk + 128, 0, 1);
    __builtin_prefetch(B + (size_t)(n0 + ml) * NDIM + kk + 128, 0, 1);
    v16bf a0 = load_frag(A + (size_t)(m0      + ml) * NDIM + kk + 8 * h);
    v16bf a1 = load_frag(A + (size_t)(m0 + 16 + ml) * NDIM + kk + 8 * h);
    #pragma unroll
    for (int j = 0; j < 4; j++) {
      v16bf b = load_frag(B + (size_t)(n0 + j * 16 + ml) * NDIM + kk + 8 * h);
      acc[0][j] = wmma_bf16(a0, b, acc[0][j]);
      acc[1][j] = wmma_bf16(a1, b, acc[1][j]);
    }
  }

  unsigned short* out = op + (size_t)c * NN;
  #pragma unroll
  for (int mi = 0; mi < 2; mi++)
    #pragma unroll
    for (int j = 0; j < 4; j++)
      #pragma unroll
      for (int g = 0; g < 8; g++)
        out[(size_t)(m0 + mi * 16 + g + 8 * h) * NDIM + n0 + j * 16 + ml] =
            f2bfbits(acc[mi][j][g]);
}

// ===================== Kernel 3: LN2 + w2 + bias + gate =====================
__global__ __launch_bounds__(256) void k3_out(
    const unsigned short* __restrict__ op, const unsigned short* __restrict__ g3w,
    const float* __restrict__ n2g, const float* __restrict__ n2b,
    const float* __restrict__ w2,  const float* __restrict__ b2,
    float* __restrict__ out)
{
  __shared__ unsigned short x2[64 * LDA];
  __shared__ unsigned short w2l[128 * LDA];

  const int tid = threadIdx.x;
  const int R0  = blockIdx.x * 64;

  // stage out_pre (channel-major) -> LDS [row][c]; coalesced along rows
  for (int idx = tid; idx < 64 * 128; idx += 256) {
    int r = idx & 63, c = idx >> 6;
    x2[r * LDA + c] = op[(size_t)c * NN + R0 + r];
  }
  __syncthreads();

  // LayerNorm per row (4 threads/row)
  {
    const int r = tid >> 2, q = tid & 3;
    float v[32];
    float s = 0.f, s2 = 0.f;
    #pragma unroll
    for (int u = 0; u < 32; u++) {
      v[u] = bf2f(x2[r * LDA + q * 32 + u]);
      s += v[u]; s2 += v[u] * v[u];
    }
    s = red4(s); s2 = red4(s2);
    const float mean = s * (1.0f / 128.0f);
    const float var  = s2 * (1.0f / 128.0f) - mean * mean;
    const float rstd = rsqrtf(var + 1e-5f);
    #pragma unroll
    for (int u = 0; u < 32; u++) {
      int c = q * 32 + u;
      float y = (v[u] - mean) * rstd * n2g[c] + n2b[c];
      x2[r * LDA + c] = f2bfbits(y);
    }
  }
  __syncthreads();

  stage_w(w2, w2l, tid);

  const int w = tid >> 5, lane = tid & 31;
  const int mt = w & 3, nh = w >> 2;
  const int ml = lane & 15, h = lane >> 4;

  v8f acc[4];
  #pragma unroll
  for (int j = 0; j < 4; j++) acc[j] = zero8();
  gemm4(x2, w2l, mt, nh, lane, acc);

  #pragma unroll
  for (int j = 0; j < 4; j++) {
    const int ng = (nh * 4 + j) * 16 + ml;
    const float bb = b2[ng];
    #pragma unroll
    for (int g = 0; g < 8; g++) {
      int R = R0 + mt * 16 + g + 8 * h;
      float gate = bf2f(g3w[(size_t)R * CDIM + ng]);
      out[(size_t)R * CDIM + ng] = (acc[j][g] + bb) * gate;
    }
  }
}

// ===================== host launch =====================
extern "C" void kernel_launch(void* const* d_in, const int* in_sizes, int n_in,
                              void* d_out, int out_size, void* d_ws, size_t ws_size,
                              hipStream_t stream) {
  const float* x    = (const float*)d_in[0];
  const float* n1g  = (const float*)d_in[1];
  const float* n1b  = (const float*)d_in[2];
  const float* n2g  = (const float*)d_in[3];
  const float* n2b  = (const float*)d_in[4];
  const float* w1i  = (const float*)d_in[5];
  const float* b1i  = (const float*)d_in[6];
  const float* w1j  = (const float*)d_in[7];
  const float* b1j  = (const float*)d_in[8];
  const float* w1is = (const float*)d_in[9];
  const float* b1is = (const float*)d_in[10];
  const float* w1js = (const float*)d_in[11];
  const float* b1js = (const float*)d_in[12];
  const float* w2   = (const float*)d_in[13];
  const float* b2   = (const float*)d_in[14];
  const float* w3   = (const float*)d_in[15];
  const float* b3   = (const float*)d_in[16];

  char* ws = (char*)d_ws;
  const size_t PLANE = (size_t)NN * CDIM * 2;   // one bf16 [c][N][N] tensor
  unsigned short* i_ws  = (unsigned short*)(ws);
  unsigned short* j_ws  = (unsigned short*)(ws + PLANE);
  unsigned short* g3_ws = (unsigned short*)(ws + 2 * PLANE);
  unsigned short* opre  = (unsigned short*)(ws + 3 * PLANE);

  k1_proj<<<NN / 64, 256, 0, stream>>>(x, n1g, n1b, w1i, b1i, w1j, b1j,
                                       w1is, b1is, w1js, b1js, w3, b3,
                                       i_ws, j_ws, g3_ws);
  k2_tri<<<CDIM * 9, 256, 0, stream>>>(i_ws, j_ws, opre);
  k3_out<<<NN / 64, 256, 0, stream>>>(opre, g3_ws, n2g, n2b, w2, b2,
                                      (float*)d_out);
}
